// TPCNNBranchPredictor_89824946029269
// MI455X (gfx1250) — matrix operand order repaired and checked by reading, not compiled, not run on hardware
//
#include <hip/hip_runtime.h>
#include <hip/hip_bf16.h>
#include <math.h>

#define TABLE 2048
#define HIST  200
#define NF    32
#define BATCH 128
#define EPS   1e-5f
#define NTILE 13              // ceil(200/16)

typedef _Float16 h16v __attribute__((ext_vector_type(16)));
typedef _Float16 h8v  __attribute__((ext_vector_type(8)));
typedef float    f8v  __attribute__((ext_vector_type(8)));

// ---------------- helpers ----------------
__device__ inline float wave_reduce_sum(float v) {
#pragma unroll
    for (int off = 16; off > 0; off >>= 1) v += __shfl_down(v, off, 32);
    return v;
}

// ---------------- kernel 1: ternary quantize weights ----------------
__global__ void k_quant(const float* __restrict__ conv_w,
                        const float* __restrict__ fc_w,
                        const float* __restrict__ q,
                        _Float16* __restrict__ cw16,
                        float* __restrict__ fwq) {
    int i = blockIdx.x * blockDim.x + threadIdx.x;
    float qv = fabsf(q[0]);
    if (i < NF * TABLE) {
        float w = conv_w[i];
        float t = (w > qv) ? 1.0f : ((w < -qv) ? -1.0f : 0.0f);
        cw16[i] = (_Float16)t;                      // exact in f16
    }
    int j = i - NF * TABLE;
    if (j >= 0 && j < NF * HIST) {
        float w = fc_w[j];
        fwq[j] = (w > qv) ? 1.0f : ((w < -qv) ? -1.0f : 0.0f);
    }
}

// ---------------- kernel 2: WMMA GEMM y[b,f,h] = sum_t cw[f,t] * x[b,t,h] ----
// One wave computes a 32x16 output tile (both f-halves) for one (b, h-tile),
// so x is streamed from HBM exactly once (memory-bound optimum).
__device__ inline h16v load_a_frag(const _Float16* __restrict__ p) {
    // halves 0..7 = K[koff .. koff+7], halves 8..15 = K[16+koff .. 16+koff+7]
    h8v lo = *reinterpret_cast<const h8v*>(p);
    h8v hi = *reinterpret_cast<const h8v*>(p + 16);
    h16v r;
#pragma unroll
    for (int j = 0; j < 8; ++j) { r[j] = lo[j]; r[j + 8] = hi[j]; }
    return r;
}

__global__ void __launch_bounds__(128)
k_gemm(const float* __restrict__ x,
       const _Float16* __restrict__ cw16,
       float* __restrict__ y) {
    const int wave = blockIdx.x * (blockDim.x >> 5) + (threadIdx.x >> 5);
    if (wave >= BATCH * NTILE) return;              // whole-wave uniform exit

    const int b  = wave / NTILE;
    const int ht = wave % NTILE;
    const int h0 = ht * 16;

    const int lane    = threadIdx.x & 31;
    const int laneMod = lane & 15;
    const int laneHi  = lane >> 4;

    // A fragment addressing (16-bit A 16x32, ISA 7.12.2)
    const int koffA = laneHi * 8;
    const _Float16* pa0 = cw16 + (size_t)laneMod * TABLE + koffA;        // rows 0..15
    const _Float16* pa1 = cw16 + (size_t)(laneMod + 16) * TABLE + koffA; // rows 16..31

    // B fragment addressing: lane holds col N=laneMod, K[koff..koff+7] and K[16+koff..16+koff+7].
    // Out-of-range columns only pollute output columns we never store, so
    // CLAMP the address (keeps loads in-bounds) and load UNCONDITIONALLY —
    // no per-lane exec masking in the hot loop.
    const int  hcol  = h0 + laneMod;
    const bool hval  = (hcol < HIST);
    const int  hcolC = hval ? hcol : (HIST - 1);
    const int  koffB = laneHi * 8;
    const float* pbBase = x + (size_t)b * TABLE * HIST + (size_t)koffB * HIST + hcolC;

    f8v acc0 = {};
    f8v acc1 = {};

#pragma unroll 2
    for (int k0 = 0; k0 < TABLE; k0 += 32) {
        h16v afrag0 = load_a_frag(pa0 + k0);
        h16v afrag1 = load_a_frag(pa1 + k0);

        const float* pb = pbBase + (size_t)k0 * HIST;
        // Speculative prefetch of next k-slab (safe past end: dropped on xlat fail)
        __builtin_prefetch((const void*)(pb + (size_t)32 * HIST), 0, 1);

        h16v bfrag;
#pragma unroll
        for (int j = 0; j < 16; ++j)
            bfrag[j] = (_Float16)pb[(size_t)(j + (j >= 8 ? 8 : 0)) * HIST];   // unconditional, clause-able

        // D = A*B + C  (f32 accumulate); args: neg_a,A,neg_b,B,c_mod,C,reuseA,reuseB
        acc0 = __builtin_amdgcn_wmma_f32_16x16x32_f16(false, afrag0, false, bfrag,
                                                      (short)0, acc0, false, false);
        acc1 = __builtin_amdgcn_wmma_f32_16x16x32_f16(false, afrag1, false, bfrag,
                                                      (short)0, acc1, false, false);
    }

    // C/D layout: lane L, VGPR v -> (M = v + 8*(L/16), N = L%16)
    if (hval) {
#pragma unroll
        for (int v = 0; v < 8; ++v) {
            int row = v + 8 * laneHi;
            y[((size_t)b * NF + row)      * HIST + hcol] = acc0[v];
            y[((size_t)b * NF + 16 + row) * HIST + hcol] = acc1[v];
        }
    }
}

// ---------------- kernel 3: BN2d batch stats per channel ----------------
__global__ void k_bnstats(const float* __restrict__ y,
                          float* __restrict__ mean,
                          float* __restrict__ rstd) {
    const int f = blockIdx.x;                       // 0..31
    float s = 0.0f, ss = 0.0f;
    for (int i = threadIdx.x; i < BATCH * HIST; i += blockDim.x) {
        int b = i / HIST, h = i % HIST;
        float v = y[((size_t)b * NF + f) * HIST + h];
        s += v; ss += v * v;
    }
    __shared__ float sm_s[8], sm_ss[8];
    s  = wave_reduce_sum(s);
    ss = wave_reduce_sum(ss);
    int wid = threadIdx.x >> 5;
    if ((threadIdx.x & 31) == 0) { sm_s[wid] = s; sm_ss[wid] = ss; }
    __syncthreads();
    if (threadIdx.x == 0) {
        float ts = 0.0f, tss = 0.0f;
        int nw = blockDim.x >> 5;
        for (int i = 0; i < nw; ++i) { ts += sm_s[i]; tss += sm_ss[i]; }
        float m = ts * (1.0f / (BATCH * HIST));
        float var = tss * (1.0f / (BATCH * HIST)) - m * m;
        mean[f] = m;
        rstd[f] = rsqrtf(var + EPS);
    }
}

// ---------------- kernel 4: BN + tanh + FC dot per batch ----------------
__global__ void k_fc(const float* __restrict__ y,
                     const float* __restrict__ mean,
                     const float* __restrict__ rstd,
                     const float* __restrict__ gamma1,
                     const float* __restrict__ beta1,
                     const float* __restrict__ fwq,
                     float* __restrict__ z) {
    const int b = blockIdx.x;                       // 0..127
    float acc = 0.0f;
    for (int i = threadIdx.x; i < NF * HIST; i += blockDim.x) {
        int f = i / HIST;                           // flat order matches reshape
        float v = y[(size_t)b * NF * HIST + i];
        float n = gamma1[f] * (v - mean[f]) * rstd[f] + beta1[f];
        acc += tanhf(n) * fwq[i];
    }
    __shared__ float sm[8];
    acc = wave_reduce_sum(acc);
    int wid = threadIdx.x >> 5;
    if ((threadIdx.x & 31) == 0) sm[wid] = acc;
    __syncthreads();
    if (threadIdx.x == 0) {
        float t = 0.0f;
        int nw = blockDim.x >> 5;
        for (int i = 0; i < nw; ++i) t += sm[i];
        z[b] = t;
    }
}

// ---------------- kernel 5: BN1d over batch + sigmoid ----------------
__global__ void k_final(const float* __restrict__ z,
                        const float* __restrict__ gamma2,
                        const float* __restrict__ beta2,
                        float* __restrict__ out) {
    const int b = threadIdx.x;                      // 128 threads, 4 waves
    float v = z[b];
    __shared__ float sm_s[4], sm_ss[4];
    float s  = wave_reduce_sum(v);
    float ss = wave_reduce_sum(v * v);
    int wid = threadIdx.x >> 5;
    if ((threadIdx.x & 31) == 0) { sm_s[wid] = s; sm_ss[wid] = ss; }
    __syncthreads();
    float ts = sm_s[0] + sm_s[1] + sm_s[2] + sm_s[3];
    float tss = sm_ss[0] + sm_ss[1] + sm_ss[2] + sm_ss[3];
    float m = ts * (1.0f / BATCH);
    float var = tss * (1.0f / BATCH) - m * m;
    float n = gamma2[0] * (v - m) * rsqrtf(var + EPS) + beta2[0];
    out[b] = 1.0f / (1.0f + expf(-n));
}

// ---------------- launcher ----------------
extern "C" void kernel_launch(void* const* d_in, const int* in_sizes, int n_in,
                              void* d_out, int out_size, void* d_ws, size_t ws_size,
                              hipStream_t stream) {
    const float* x      = (const float*)d_in[0];
    const float* conv_w = (const float*)d_in[1];
    const float* fc_w   = (const float*)d_in[2];
    const float* gamma1 = (const float*)d_in[3];
    const float* beta1  = (const float*)d_in[4];
    const float* gamma2 = (const float*)d_in[5];
    const float* beta2  = (const float*)d_in[6];
    const float* q      = (const float*)d_in[7];
    float* out          = (float*)d_out;

    // workspace layout (all offsets 16B-aligned)
    char* ws = (char*)d_ws;
    _Float16* cw16 = (_Float16*)(ws);                                   // 32*2048*2   = 131072
    float*    fwq  = (float*)(ws + 131072);                             // 6400*4      =  25600
    float*    y    = (float*)(ws + 156672);                             // 128*32*200*4= 3276800
    float*    mean = (float*)(ws + 156672 + 3276800);                   // 32*4
    float*    rstd = (float*)(ws + 156672 + 3276800 + 128);             // 32*4
    float*    z    = (float*)(ws + 156672 + 3276800 + 256);             // 128*4
    (void)in_sizes; (void)n_in; (void)out_size; (void)ws_size;

    // 1) ternary quantize weights (conv -> f16 for WMMA A, fc -> f32)
    {
        int total = NF * TABLE + NF * HIST;
        k_quant<<<(total + 255) / 256, 256, 0, stream>>>(conv_w, fc_w, q, cw16, fwq);
    }
    // 2) batched GEMM via v_wmma_f32_16x16x32_f16: 1664 waves = 128 batches x 13 h-tiles
    {
        int waves  = BATCH * NTILE;                  // 1664
        int blocks = (waves + 3) / 4;                // 4 waves (128 thr) per block
        k_gemm<<<blocks, 128, 0, stream>>>(x, cw16, y);
    }
    // 3) BN2d stats (per-channel mean/rstd over batch,hist)
    k_bnstats<<<NF, 256, 0, stream>>>(y, mean, rstd);
    // 4) BN + tanh + ternary FC reduction per batch
    k_fc<<<BATCH, 256, 0, stream>>>(y, mean, rstd, gamma1, beta1, fwq, z);
    // 5) BN1d over batch + sigmoid
    k_final<<<1, BATCH, 0, stream>>>(z, gamma2, beta2, out);
}
